// SSM_29832842838322
// MI455X (gfx1250) — matrix operand order repaired and checked by reading, not compiled, not run on hardware
//
#include <hip/hip_runtime.h>
#include <hip/hip_bf16.h>

// ---------------------------------------------------------------------------
// SSM block for MI455X (gfx1250):
//   O_lh/O_rh  : fp32 Newton-Schulz polar (V_WMMA_F32_16X16X4_F32)
//   q,v,qr,Y   : bf16 WMMA GEMMs (V_WMMA_F32_16X16X32_BF16), fp32 accum
//   GEMM tiles : GLOBAL_LOAD_ASYNC_TO_LDS_B128 + double-buffered LDS pipeline
//                (1 barrier per K-tile), fallback to load/ds_store if absent
//   scan       : fp32, 16 persistent WGs, O_lh slice resident in 320KB LDS,
//                h re-staged each step via TENSOR_LOAD_TO_LDS (TDM),
//                2 device-scope barriers per step
// ---------------------------------------------------------------------------

typedef __attribute__((ext_vector_type(2)))  float  v2f;
typedef __attribute__((ext_vector_type(8)))  float  v8f;
typedef __attribute__((ext_vector_type(16))) __bf16 v16bf;
typedef int v4i __attribute__((ext_vector_type(4)));

#define C_DIM 1024
#define B_DIM 4
#define T_DIM 4096
#define BT_DIM (B_DIM * T_DIM)      // 16384
#define SCAN_WGS 16

#ifndef __has_builtin
#define __has_builtin(x) 0
#endif
#if __has_builtin(__builtin_amdgcn_global_load_async_to_lds_b128)
#define HAVE_ASYNC_LDS 1
#else
#define HAVE_ASYNC_LDS 0
#endif
#if __has_builtin(__builtin_amdgcn_tensor_load_to_lds)
#define HAVE_TDM 1
#else
#define HAVE_TDM 0
#endif

#define AS1 __attribute__((address_space(1)))
#define AS3 __attribute__((address_space(3)))

#if HAVE_ASYNC_LDS
// copies 16B from per-lane global addr to per-lane LDS addr; imm offset applies
// to BOTH addresses (ISA 15.18.3), so contiguous 32B needs imm 0 and 16.
// Param types per clang-23: (v4i addrspace(1)*, v4i addrspace(3)*, Ii, Ii).
#define ASYNC_B128(g, l, imm)                                                  \
    __builtin_amdgcn_global_load_async_to_lds_b128(                            \
        (AS1 v4i*)(AS1 void*)(void*)(g), (AS3 v4i*)(AS3 void*)(l), (imm), 0)
#endif

__device__ inline void wait_async0() {
#if __has_builtin(__builtin_amdgcn_s_wait_asynccnt)
    __builtin_amdgcn_s_wait_asynccnt(0);
#elif HAVE_ASYNC_LDS
    asm volatile("s_wait_asynccnt 0x0" ::: "memory");
#endif
}

__device__ inline void wait_tensor0() {
#if __has_builtin(__builtin_amdgcn_s_wait_tensorcnt)
    __builtin_amdgcn_s_wait_tensorcnt(0);
#elif HAVE_TDM
    asm volatile("s_wait_tensorcnt 0x0" ::: "memory");
#endif
}

#if HAVE_TDM
typedef unsigned int u32x4 __attribute__((ext_vector_type(4)));
typedef int          i32x8 __attribute__((ext_vector_type(8)));
typedef int          i32x4 __attribute__((ext_vector_type(4)));

// TDM 2D tile load: `rows` rows of `elems` f32 each, row stride `stride_elems`,
// packed contiguously at LDS byte offset `lds_off`. D# per ISA ch.8 (CDNA5).
// clang-23 builtin arity: (u32x4 g0, i32x8 g1, i32x4 g2, i32x4 g3, i32x8, i32 cpol)
__device__ inline void tdm_load_rows_f32(unsigned lds_off, const void* gsrc,
                                         unsigned elems, unsigned rows,
                                         unsigned stride_elems) {
    unsigned long long ga = (unsigned long long)gsrc;
    u32x4 g0;
    g0[0] = 1u;                                            // count=1, user mode
    g0[1] = lds_off;                                       // lds_addr (bytes)
    g0[2] = (unsigned)ga;                                  // global_addr[31:0]
    g0[3] = (unsigned)((ga >> 32) & 0x01ffffffu) | (2u << 30); // [56:32] | type=2
    i32x8 g1;
    g1[0] = (int)(2u << 16);                               // data_size=2 -> 4B
    g1[1] = (int)((elems & 0xffffu) << 16);                // tensor_dim0[15:0]
    g1[2] = (int)((elems >> 16) | ((rows & 0xffffu) << 16));   // d0 hi | d1 lo
    g1[3] = (int)((rows >> 16) | ((elems & 0xffffu) << 16));   // d1 hi | tile_dim0
    g1[4] = (int)(rows & 0xffffu);                         // tile_dim1 (tile_dim2=0)
    g1[5] = (int)stride_elems;                             // tensor_dim0_stride lo
    g1[6] = 0;
    g1[7] = 0;
    i32x4 z4 = {0, 0, 0, 0};                               // groups 2/3: dims 0
    i32x8 z8 = {0, 0, 0, 0, 0, 0, 0, 0};
    __builtin_amdgcn_tensor_load_to_lds(g0, g1, z4, z4, z8, 0);
}
#endif

// ------------------------------ small utility kernels ----------------------

__global__ void reduce_sumsq_kernel(const float* __restrict__ w, float* acc, int n) {
    __shared__ float red[256];
    float s = 0.f;
    for (int i = blockIdx.x * 256 + threadIdx.x; i < n; i += gridDim.x * 256) {
        float v = w[i]; s += v * v;
    }
    red[threadIdx.x] = s; __syncthreads();
    for (int st = 128; st > 0; st >>= 1) {
        if (threadIdx.x < st) red[threadIdx.x] += red[threadIdx.x + st];
        __syncthreads();
    }
    if (threadIdx.x == 0) atomicAdd(acc, red[0]);
}

__global__ void scale_by_invnorm_kernel(const float* __restrict__ w,
                                        const float* __restrict__ sumsq,
                                        float* __restrict__ x, int n) {
    float inv = 1.f / (sqrtf(*sumsq) + 1e-8f);
    int i = blockIdx.x * 256 + threadIdx.x;
    if (i < n) x[i] = w[i] * inv;
}

__global__ void f32_to_bf16_kernel(const float* __restrict__ in,
                                   __bf16* __restrict__ out, int n) {
    int i = blockIdx.x * 256 + threadIdx.x;
    if (i < n) out[i] = (__bf16)in[i];
}

__global__ void mul_to_bf16_kernel(const float* __restrict__ a,
                                   const __bf16* __restrict__ b,
                                   __bf16* __restrict__ out, int n) {
    int i = blockIdx.x * 256 + threadIdx.x;
    if (i < n) out[i] = (__bf16)(a[i] * (float)b[i]);
}

__global__ void init_h_kernel(const float* __restrict__ identity,
                              float* __restrict__ h) {
    int i = blockIdx.x * 256 + threadIdx.x;   // B_DIM*C_DIM threads
    h[i] = identity[i & (C_DIM - 1)];
}

// ------------------------------ fp32 WMMA GEMM -----------------------------
// C[M,N] = alpha * (A op(B)) + beta * Z     contraction over K
//   TRANS_B=1: op(B)[k][n] = Bm[n*K + k]   (NT, used for X @ X^T)
//   TRANS_B=0: op(B)[k][n] = Bm[k*N + n]   (NN, used for (XX^T) @ X)
// Block tile 128x64, 8 waves (4x2), each wave 2x2 tiles of 16x16, KT=16.
template <int TRANS_B>
__global__ __launch_bounds__(256) void gemm_f32_wmma(
    const float* __restrict__ A, const float* __restrict__ Bm,
    float* __restrict__ Cout, const float* __restrict__ Z,
    float alpha, float beta, int M, int N, int K) {
    __shared__ float Ash[2][128][20];
    __shared__ float Bnt[2][64][20];
    __shared__ float Bnn[2][16][68];

    const int tid = threadIdx.x;
    const int lane = tid & 31, wid = tid >> 5;
    const int wm = wid & 3, wn = wid >> 2;
    const int lm = lane & 15, kh = lane >> 4;
    const int bm0 = blockIdx.x * 128, bn0 = blockIdx.y * 64;

    const int rA = tid >> 1, chA = tid & 1;           // 8 f32 each (32B)
    const int rBt = tid >> 2, chBt = tid & 3;         // NT: 4 f32 each
    const int rBn = tid >> 4, chBn = tid & 15;        // NN: 4 f32 each
    const float* gA  = A + (size_t)(bm0 + rA) * K + chA * 8;
    const float* gBt = Bm + (size_t)(bn0 + rBt) * K + chBt * 4;
    const float* gBn = Bm + (size_t)rBn * N + bn0 + chBn * 4;

    v8f acc[2][2] = {};

    auto compute = [&](int bs) {
#pragma unroll
        for (int k4 = 0; k4 < 4; ++k4) {
            const int kb = k4 * 4 + kh * 2;
#pragma unroll
            for (int mi = 0; mi < 2; ++mi) {
                const int ra = wm * 32 + mi * 16 + lm;
                v2f a = { Ash[bs][ra][kb], Ash[bs][ra][kb + 1] };
#pragma unroll
                for (int ni = 0; ni < 2; ++ni) {
                    const int rb = wn * 32 + ni * 16 + lm;
                    v2f b;
                    if (TRANS_B) b = (v2f){ Bnt[bs][rb][kb], Bnt[bs][rb][kb + 1] };
                    else         b = (v2f){ Bnn[bs][kb][rb], Bnn[bs][kb + 1][rb] };
                    acc[mi][ni] = __builtin_amdgcn_wmma_f32_16x16x4_f32(
                        false, a, false, b, (short)0, acc[mi][ni], false, false);
                }
            }
        }
    };

#if HAVE_ASYNC_LDS
    auto issue = [&](int kt, int bs) {
        ASYNC_B128(gA + kt, &Ash[bs][rA][chA * 8], 0);
        ASYNC_B128(gA + kt, &Ash[bs][rA][chA * 8], 16);
        if (TRANS_B) ASYNC_B128(gBt + kt, &Bnt[bs][rBt][chBt * 4], 0);
        else         ASYNC_B128(gBn + (size_t)kt * N, &Bnn[bs][rBn][chBn * 4], 0);
    };
    const int NT = K >> 4;
    issue(0, 0);
    for (int i = 0; i < NT; ++i) {
        wait_async0();                // own wave's async loads done
        __syncthreads();              // everyone's done; prev-prev buffer free
        if (i + 1 < NT) issue((i + 1) << 4, (i + 1) & 1);
        compute(i & 1);
    }
#else
    for (int kt = 0; kt < K; kt += 16) {
        if (kt + 16 < K) __builtin_prefetch((const void*)(gA + kt + 16), 0, 0);
        *(float4*)&Ash[0][rA][chA * 8]     = *(const float4*)(gA + kt);
        *(float4*)&Ash[0][rA][chA * 8 + 4] = *(const float4*)(gA + kt + 4);
        if (TRANS_B) *(float4*)&Bnt[0][rBt][chBt * 4] = *(const float4*)(gBt + kt);
        else         *(float4*)&Bnn[0][rBn][chBn * 4] = *(const float4*)(gBn + (size_t)kt * N);
        __syncthreads();
        compute(0);
        __syncthreads();
    }
#endif

#pragma unroll
    for (int mi = 0; mi < 2; ++mi)
#pragma unroll
        for (int ni = 0; ni < 2; ++ni) {
            const int mrow = bm0 + wm * 32 + mi * 16 + kh * 8;
            const int ncol = bn0 + wn * 32 + ni * 16 + lm;
#pragma unroll
            for (int r = 0; r < 8; ++r) {
                size_t idx = (size_t)(mrow + r) * N + ncol;
                float vv = alpha * acc[mi][ni][r];
                if (Z) vv += beta * Z[idx];
                Cout[idx] = vv;
            }
        }
}

// ------------------------------ bf16 WMMA GEMM (NT) ------------------------
// C[M,N] = A[M,K] * Bw[N,K]^T  (nn.Linear layout), fp32 accumulate.
// Block tile 128x64, 8 waves (4x2), 2x2 WMMA tiles each, KT=32.
template <int STORE_BF16>
__global__ __launch_bounds__(256) void gemm_bf16_wmma(
    const __bf16* __restrict__ A, const __bf16* __restrict__ Bw,
    float* __restrict__ Cf, __bf16* __restrict__ Cb, int M, int N, int K) {
    __shared__ __bf16 Ash[2][128][40];
    __shared__ __bf16 Bsh[2][64][40];

    const int tid = threadIdx.x;
    const int lane = tid & 31, wid = tid >> 5;
    const int wm = wid & 3, wn = wid >> 2;
    const int lm = lane & 15, kh = lane >> 4;
    const int bm0 = blockIdx.x * 128, bn0 = blockIdx.y * 64;

    const int rA = tid >> 1, chA = tid & 1;           // 16 bf16 each (32B)
    const int rB = tid >> 2, chB = tid & 3;           // 8 bf16 each (16B)
    const __bf16* gA = A + (size_t)(bm0 + rA) * K + chA * 16;
    const __bf16* gB = Bw + (size_t)(bn0 + rB) * K + chB * 8;

    v8f acc[2][2] = {};

    auto compute = [&](int bs) {
#pragma unroll
        for (int mi = 0; mi < 2; ++mi) {
            const int ra = wm * 32 + mi * 16 + lm;
            v16bf a;   // A 16x32: lanes 0-15 K=0..7/16..23, lanes 16-31 K=8..15/24..31
#pragma unroll
            for (int e = 0; e < 8; ++e) {
                a[e]     = Ash[bs][ra][kh * 8 + e];
                a[e + 8] = Ash[bs][ra][16 + kh * 8 + e];
            }
#pragma unroll
            for (int ni = 0; ni < 2; ++ni) {
                const int rb = wn * 32 + ni * 16 + lm;
                v16bf b;  // B 32x16: lanes 0-15 K=0..7/16..23, lanes 16-31 K=8..15/24..31
#pragma unroll
                for (int e = 0; e < 16; ++e) b[e] = Bsh[bs][rb][(e < 8 ? kh * 8 + e : 16 + kh * 8 + (e - 8))];
                acc[mi][ni] = __builtin_amdgcn_wmma_f32_16x16x32_bf16(
                    false, a, false, b, (short)0, acc[mi][ni], false, false);
            }
        }
    };

#if HAVE_ASYNC_LDS
    auto issue = [&](int kt, int bs) {
        ASYNC_B128(gA + kt, &Ash[bs][rA][chA * 16], 0);
        ASYNC_B128(gA + kt, &Ash[bs][rA][chA * 16], 16);
        ASYNC_B128(gB + kt, &Bsh[bs][rB][chB * 8], 0);
    };
    const int NT = K >> 5;
    issue(0, 0);
    for (int i = 0; i < NT; ++i) {
        wait_async0();
        __syncthreads();
        if (i + 1 < NT) issue((i + 1) << 5, (i + 1) & 1);
        compute(i & 1);
    }
#else
    for (int kt = 0; kt < K; kt += 32) {
        if (kt + 32 < K) __builtin_prefetch((const void*)(gA + kt + 32), 0, 0);
        *(uint4*)&Ash[0][rA][chA * 16]     = *(const uint4*)(gA + kt);
        *(uint4*)&Ash[0][rA][chA * 16 + 8] = *(const uint4*)(gA + kt + 8);
        *(uint4*)&Bsh[0][rB][chB * 8]      = *(const uint4*)(gB + kt);
        __syncthreads();
        compute(0);
        __syncthreads();
    }
#endif

#pragma unroll
    for (int mi = 0; mi < 2; ++mi)
#pragma unroll
        for (int ni = 0; ni < 2; ++ni) {
            const int mrow = bm0 + wm * 32 + mi * 16 + kh * 8;
            const int ncol = bn0 + wn * 32 + ni * 16 + lm;
#pragma unroll
            for (int r = 0; r < 8; ++r) {
                size_t idx = (size_t)(mrow + r) * N + ncol;
                if (STORE_BF16) Cb[idx] = (__bf16)acc[mi][ni][r];
                else            Cf[idx] = acc[mi][ni][r];
            }
        }
}

// ------------------------------ persistent scan ----------------------------

__device__ inline void gbar(int* ctr, int id, int nwg) {
    __threadfence();
    __syncthreads();
    if (threadIdx.x == 0) {
        __hip_atomic_fetch_add(&ctr[id], 1, __ATOMIC_RELEASE, __HIP_MEMORY_SCOPE_AGENT);
        while (__hip_atomic_load(&ctr[id], __ATOMIC_ACQUIRE, __HIP_MEMORY_SCOPE_AGENT) < nwg)
            __builtin_amdgcn_s_sleep(2);
    }
    __syncthreads();
    __threadfence();
}

// 16 WGs x 256 threads. WG g owns output columns d in [64g, 64g+64).
// Its 64x1024 slice of O_lh lives (transposed) in LDS for the whole scan.
// h (16KB) is re-staged into LDS each step via the Tensor Data Mover.
__global__ __launch_bounds__(256) void scan_kernel(
    const float* __restrict__ Olh,      // [C][C]  O_lh[d][c]
    const float* __restrict__ qr,       // [B][T][C]  q @ O_rh^T
    float* __restrict__ h,              // [B][C] global h (initialized)
    float* __restrict__ partials,       // [SCAN_WGS * B]
    int* __restrict__ ctr,              // [2*T] zeroed counters
    float* __restrict__ qs) {           // [B][T][C] output
    extern __shared__ float smem[];
    float* Ot  = smem;                       // [C][64]  Ot[c][dl] = O_lh[g*64+dl][c]
    float* hsh = Ot + (size_t)C_DIM * 64;    // [B*C]
    float* red = hsh + B_DIM * C_DIM;        // [256]

    const int tid = threadIdx.x, g = blockIdx.x;
    const int b = tid >> 6, dl = tid & 63;
    const int dglob = g * 64 + dl;

    { // stage O slice (coalesced global reads, transposed LDS writes)
        const int drow = tid >> 2, cpart = tid & 3;
        const float* src = Olh + (size_t)(g * 64 + drow) * C_DIM + cpart * 256;
        for (int c = 0; c < 256; ++c)
            Ot[(size_t)(cpart * 256 + c) * 64 + drow] = src[c];
    }

    auto stage_h = [&]() {
#if HAVE_TDM
        if (tid < 32) {   // wave 0 issues the DMA (scalar op), waits TENSORcnt
            tdm_load_rows_f32((unsigned)(unsigned long long)(AS3 void*)hsh,
                              (const void*)h, B_DIM * C_DIM, 1, B_DIM * C_DIM);
            wait_tensor0();
        }
#else
        for (int j = 0; j < 16; ++j) hsh[tid * 16 + j] = h[tid * 16 + j];
#endif
    };

    stage_h();
    __syncthreads();

    for (int t = 0; t < T_DIM; ++t) {
        // y[b][dglob] = qr[b][t][dglob] + sum_c h[b][c] * O_lh[dglob][c]
        float acc = qr[((size_t)b * T_DIM + t) * C_DIM + dglob];
        const float* hb = hsh + b * C_DIM;
        const float* oc = Ot + dl;
#pragma unroll 8
        for (int c = 0; c < C_DIM; ++c) acc += hb[c] * oc[(size_t)c * 64];

        // local sum of squares over this WG's 64 columns, per batch row
        red[tid] = acc * acc;
        __syncthreads();
        for (int s = 32; s > 0; s >>= 1) {
            if (dl < s) red[tid] += red[tid + s];
            __syncthreads();
        }
        if (dl == 0) partials[g * B_DIM + b] = red[tid];

        gbar(ctr, 2 * t, SCAN_WGS);

        float s = 0.f;
        for (int i = 0; i < SCAN_WGS; ++i) s += partials[i * B_DIM + b];
        const float scale = rsqrtf(s * (1.0f / C_DIM) + 1e-6f);
        const float hn = acc * scale;
        h[(size_t)b * C_DIM + dglob] = hn;
        qs[((size_t)b * T_DIM + t) * C_DIM + dglob] = hn;

        gbar(ctr, 2 * t + 1, SCAN_WGS);

        stage_h();
        __syncthreads();
    }
}

// ------------------------------ host orchestration -------------------------

extern "C" void kernel_launch(void* const* d_in, const int* in_sizes, int n_in,
                              void* d_out, int out_size, void* d_ws, size_t ws_size,
                              hipStream_t stream) {
    const float* x_f   = (const float*)d_in[0];   // [4,4096,1024]
    const float* Wq_f  = (const float*)d_in[1];   // [1024,1024]
    const float* Wv_f  = (const float*)d_in[2];
    const float* Wc_f  = (const float*)d_in[3];
    const float* ident = (const float*)d_in[4];   // [1024]
    const float* Wlh_f = (const float*)d_in[5];
    const float* Wrh_f = (const float*)d_in[6];
    float* Y_out  = (float*)d_out;                          // [4,4096,1024]
    float* qs_out = (float*)d_out + (size_t)BT_DIM * C_DIM; // [4,4096,1024]

    const int CC = C_DIM * C_DIM;          // 1048576
    const int NX = BT_DIM * C_DIM;         // 16777216

    // workspace carve-up
    char* w = (char*)d_ws; size_t off = 0;
    auto carve = [&](size_t bytes) -> void* {
        void* p = w + off; off += (bytes + 255) & ~(size_t)255; return p;
    };
    float*  P0     = (float*)carve((size_t)CC * 4);     // O_lh (final)
    float*  P1     = (float*)carve((size_t)CC * 4);     // NS ping-pong scratch
    float*  P2     = (float*)carve((size_t)CC * 4);     // O_rh (final)
    float*  P3     = (float*)carve((size_t)CC * 4);     // A = X X^T scratch
    __bf16* xbf    = (__bf16*)carve((size_t)NX * 2);
    __bf16* qbf    = (__bf16*)carve((size_t)NX * 2);
    __bf16* vbf    = (__bf16*)carve((size_t)NX * 2);
    __bf16* ubf    = (__bf16*)carve((size_t)NX * 2);
    float*  qr     = (float*)carve((size_t)NX * 4);
    __bf16* Wqb    = (__bf16*)carve((size_t)CC * 2);
    __bf16* Wvb    = (__bf16*)carve((size_t)CC * 2);
    __bf16* Wcb    = (__bf16*)carve((size_t)CC * 2);
    __bf16* Orhb   = (__bf16*)carve((size_t)CC * 2);
    float*  hbuf   = (float*)carve((size_t)B_DIM * C_DIM * 4);
    float*  parts  = (float*)carve((size_t)SCAN_WGS * B_DIM * 4);
    int*    ctr    = (int*)carve((size_t)2 * T_DIM * 4);
    float*  norm   = (float*)carve(4);

    const dim3 thr(256);
    const dim3 gSq(8, 16);     // 1024x1024 GEMM (128x64 tiles)
    const dim3 gBig(128, 16);  // 16384x1024 GEMM

    // ---- polar orthogonalization (fp32 WMMA Newton-Schulz) ----
    auto polar = [&](const float* Win, float* pa, float* pb) {
        (void)hipMemsetAsync(norm, 0, 4, stream);
        reduce_sumsq_kernel<<<1024, thr, 0, stream>>>(Win, norm, CC);
        scale_by_invnorm_kernel<<<CC / 256, thr, 0, stream>>>(Win, norm, pa, CC);
        for (int it = 0; it < 12; ++it) {
            // P3 = X X^T  (NT)
            gemm_f32_wmma<1><<<gSq, thr, 0, stream>>>(pa, pa, P3, (const float*)nullptr,
                                                      1.0f, 0.0f, C_DIM, C_DIM, C_DIM);
            // pb = 1.5*X - 0.5*(P3 @ X)  (NN with fused epilogue)
            gemm_f32_wmma<0><<<gSq, thr, 0, stream>>>(P3, pa, pb, pa,
                                                      -0.5f, 1.5f, C_DIM, C_DIM, C_DIM);
            float* t = pa; pa = pb; pb = t;
        }
        // 12 iterations (even) -> result back in the original pa buffer
    };
    polar(Wlh_f, P0, P1);   // O_lh in P0
    polar(Wrh_f, P2, P1);   // O_rh in P2

    // ---- precision conversions ----
    f32_to_bf16_kernel<<<NX / 256, thr, 0, stream>>>(x_f, xbf, NX);
    f32_to_bf16_kernel<<<CC / 256, thr, 0, stream>>>(Wq_f, Wqb, CC);
    f32_to_bf16_kernel<<<CC / 256, thr, 0, stream>>>(Wv_f, Wvb, CC);
    f32_to_bf16_kernel<<<CC / 256, thr, 0, stream>>>(Wc_f, Wcb, CC);
    f32_to_bf16_kernel<<<CC / 256, thr, 0, stream>>>(P2, Orhb, CC);

    // ---- projections: q = x Wq^T (bf16 out), v = x Wv^T (bf16 out) ----
    gemm_bf16_wmma<1><<<gBig, thr, 0, stream>>>(xbf, Wqb, (float*)nullptr, qbf,
                                                BT_DIM, C_DIM, C_DIM);
    gemm_bf16_wmma<1><<<gBig, thr, 0, stream>>>(xbf, Wvb, (float*)nullptr, vbf,
                                                BT_DIM, C_DIM, C_DIM);
    // qr = q O_rh^T  (fp32 out; feeds the recurrence)
    gemm_bf16_wmma<0><<<gBig, thr, 0, stream>>>(qbf, Orhb, qr, (__bf16*)nullptr,
                                                BT_DIM, C_DIM, C_DIM);

    // ---- sequential scan, parallel over 16 WGPs ----
    init_h_kernel<<<(B_DIM * C_DIM) / 256, thr, 0, stream>>>(ident, hbuf);
    (void)hipMemsetAsync(ctr, 0, (size_t)2 * T_DIM * 4, stream);
    const size_t scan_lds = ((size_t)C_DIM * 64 + B_DIM * C_DIM + 256) * sizeof(float);
    scan_kernel<<<SCAN_WGS, thr, scan_lds, stream>>>(P0, qr, hbuf, parts, ctr, qs_out);

    // ---- output projection: Y = (qs * v) Wc^T ----
    mul_to_bf16_kernel<<<NX / 256, thr, 0, stream>>>(qs_out, vbf, ubf, NX);
    gemm_bf16_wmma<0><<<gBig, thr, 0, stream>>>(ubf, Wcb, Y_out, (__bf16*)nullptr,
                                                BT_DIM, C_DIM, C_DIM);
}
